// FlipFlop_59493886984845
// MI455X (gfx1250) — matrix operand order repaired and checked by reading, not compiled, not run on hardware
//
#include <hip/hip_runtime.h>

// Problem constants (from reference): B=256, T=512, D=64, H=512, C=64
#define B_ 256
#define T_ 512
#define D_ 64
#define H_ 512
#define C_ 64
#define DT_ 0.1f

typedef __bf16 bf16;
typedef __attribute__((ext_vector_type(16))) __bf16 v16bf;
typedef __attribute__((ext_vector_type(8)))  __bf16 v8bf;
typedef __attribute__((ext_vector_type(8)))  float  v8f;
typedef __attribute__((ext_vector_type(4)))  float  v4f;

// Explicitly global-address-space fragment pointer: survives the anti-LICM
// asm with provenance intact -> global_load_b128 (LOADcnt only), not flat.
typedef const __attribute__((address_space(1))) v16bf gv16bf;

#if __has_builtin(__builtin_amdgcn_global_load_async_to_lds_b32) && \
    __has_builtin(__builtin_amdgcn_s_wait_asynccnt)
#define USE_ASYNC_X 1
// Pointer types per clang's diagnostic: int* in AS(1) (global) / AS(3) (LDS)
typedef __attribute__((address_space(1))) int gas_int;
typedef __attribute__((address_space(3))) int las_int;
#endif

__device__ __forceinline__ v8f wmma_bf16(v16bf a, v16bf b, v8f c) {
  // D(f32 16x16) = A(bf16 16x32) * B(bf16 32x16) + C
  return __builtin_amdgcn_wmma_f32_16x16x32_bf16(
      /*neg_a=*/false, a, /*neg_b=*/false, b,
      /*c_mod=*/(short)0, c, /*reuse_a=*/false, /*reuse_b=*/false);
}

// ---------------------------------------------------------------------------
// Pack an (N x K) fp32 weight matrix into WMMA B-fragment order, bf16.
// Result[b,h] = sum_k A[b,k] * Wt[h,k]  ->  B-operand[k][n] = Wt[n0+n, k].
// Fragment layout (ISA 7.12.2): lane L holds column n = L%16; element s holds
// k = kt*32 + 8*(L/16) + (s&7) + 16*(s>>3). Packed index: ((nt*Ktiles + kt)*32 + lane)*16 + s.
// Rows >= zero_from are zeroed (for the masked P matrix).
// ---------------------------------------------------------------------------
__global__ void pack_b_kernel(const float* __restrict__ src, bf16* __restrict__ dst,
                              int N, int K, int zero_from) {
  int id = blockIdx.x * 256 + threadIdx.x;
  if (id >= N * K) return;
  int s    = id & 15;          // element: k slot within tile
  int l    = (id >> 4) & 31;   // lane: n = l & 15, k half = l >> 4
  int tile = id >> 9;          // nt*Ktiles + kt
  int Ktiles = K >> 5;
  int kt = tile % Ktiles;
  int nt = tile / Ktiles;
  int row = nt * 16 + (l & 15);       // weight row (output feature)
  int col = kt * 32 + (l >> 4) * 8 + (s & 7) + (s >> 3) * 16;       // weight col (reduction dim)
  float v = (row < zero_from) ? src[row * K + col] : 0.0f;
  dst[id] = (bf16)v;
}

// ---------------------------------------------------------------------------
// Recurrent scan. grid = B/16 workgroups, 1024 threads (32 waves).
// Wave `nt` owns H-columns [nt*16, nt*16+16). r=relu(v) is shared via LDS.
// Ping-pong r/x buffers -> single barrier per step; x_{t+1} is staged with an
// async global->LDS copy fired before the step's WMMAs and waited just before
// the barrier, so its latency hides entirely behind the matrix pipe.
// W/K fragments stream from L2 every step (anti-LICM asm keeps the loads in
// the loop; AS(1) typing keeps them global_load_b128), software-pipelined one
// k-tile ahead of the WMMAs.
// ---------------------------------------------------------------------------
__global__ __launch_bounds__(1024)
void rnn_kernel(const float* __restrict__ x,
                const bf16* __restrict__ Wp,  const bf16* __restrict__ Kp,
                const bf16* __restrict__ Pmp, const bf16* __restrict__ Pzp,
                const float* __restrict__ bv, const float* __restrict__ bz,
                float* __restrict__ hidden)
{
  __shared__ __align__(16) bf16  rbuf[2][16 * H_];  // r ping-pong (2 x 16KB)
  __shared__ __align__(16) float xbuf[2][16 * D_];  // x_t ping-pong (2 x 4KB)

  const int tid   = threadIdx.x;
  const int lane  = tid & 31;
  const int nt    = tid >> 5;          // wave id == N tile (0..31)
  const int b0    = blockIdx.x * 16;   // batch-tile base row

  const int half  = lane >> 4;         // 0 | 1
  const int abase = half * 8;          // A-frag K sub-base; also C-frag M base
  const int n     = lane & 15;
  const int h     = nt * 16 + n;       // my output column
  const int arow  = lane & 15;         // A-frag M row

  v8f v = {0.f,0.f,0.f,0.f,0.f,0.f,0.f,0.f};   // fp32 state tile (16x16)

  const float bzl = bz[h];
  const float bvl = bv[h];

  // Per-wave, per-lane weight fragment pointers (divergent -> VGPR pairs),
  // explicitly typed as global so loads stay global_load_b128 after the asm.
  gv16bf* wfrag = (gv16bf*)((const v16bf*)Wp + (size_t)nt * (H_ / 32) * 32 + lane);
  gv16bf* kfrag = (gv16bf*)((const v16bf*)Kp + (size_t)nt * (H_ / 32) * 32 + lane);

  // Small x-projection fragments: keep register-resident for all 512 steps.
  v16bf bpm[D_ / 32], bpz[D_ / 32];
  #pragma unroll
  for (int kt = 0; kt < D_ / 32; ++kt) {
    bpm[kt] = ((const v16bf*)Pmp)[(nt * (D_ / 32) + kt) * 32 + lane];
    bpz[kt] = ((const v16bf*)Pzp)[(nt * (D_ / 32) + kt) * 32 + lane];
  }

  // A-fragment gather from an LDS r-buffer (ISA 7.12.2 16-bit A layout)
  auto load_a = [&](const bf16* rb, int kt) -> v16bf {
    const int c0 = kt * 32 + abase;
    v8bf lo = *(const v8bf*)&rb[arow * H_ + c0];
    v8bf hi = *(const v8bf*)&rb[arow * H_ + c0 + 16];
    v16bf a;
    #pragma unroll
    for (int i = 0; i < 8; ++i) { a[i] = lo[i]; a[i + 8] = hi[i]; }
    return a;
  };

  // x staging: one fp32 element per thread per step
  const int xm = tid >> 6;             // 0..15
  const int xd = tid & 63;             // 0..63
  const size_t xrow = (size_t)(b0 + xm) * T_ * D_ + xd;

  // Prologue: r0 = relu(0) = 0 into buffer 0; stage x_0 into buffer 0.
  for (int i = tid; i < 16 * H_; i += 1024) rbuf[0][i] = (bf16)0.0f;
  xbuf[0][xm * D_ + xd] = x[xrow];
  __syncthreads();

  int cur = 0;
  for (int t = 0; t < T_; ++t) {
    // Opaque per-iteration pointers (VGPR pairs -> "v" constraint): blocks
    // LICM from hoisting (and then spilling) 1KB/lane of loop-invariant
    // weight fragments.
    asm volatile("" : "+v"(wfrag));
    asm volatile("" : "+v"(kfrag));

    const int nxt = cur ^ 1;
    const bf16*  rb = rbuf[cur];
    const float* xb = xbuf[cur];

    // Fire-and-forget staging of x_{t+1} into the other buffer; latency
    // hides behind the 36 WMMAs below.
#ifdef USE_ASYNC_X
    if (t + 1 < T_) {
      __builtin_amdgcn_global_load_async_to_lds_b32(
          (gas_int*)&x[xrow + (size_t)(t + 1) * D_],
          (las_int*)&xbuf[nxt][xm * D_ + xd], 0, 0);
    }
#else
    float xstage = 0.0f;
    if (t + 1 < T_) xstage = x[xrow + (size_t)(t + 1) * D_];
#endif

    v8f accW = {0.f,0.f,0.f,0.f,0.f,0.f,0.f,0.f};
    v8f accK = {0.f,0.f,0.f,0.f,0.f,0.f,0.f,0.f};

    // r @ W^T and r @ K^T : K = 512 -> 16 k-tiles, 2 WMMAs each,
    // software-pipelined one k-tile ahead.
    v16bf a_cur  = load_a(rb, 0);
    v16bf bw_cur = wfrag[0];
    v16bf bk_cur = kfrag[0];
    #pragma unroll
    for (int kt = 0; kt < H_ / 32; ++kt) {
      v16bf a_nxt = a_cur, bw_nxt = bw_cur, bk_nxt = bk_cur;
      if (kt + 1 < H_ / 32) {
        a_nxt  = load_a(rb, kt + 1);
        bw_nxt = wfrag[(kt + 1) * 32];
        bk_nxt = kfrag[(kt + 1) * 32];
      }
      accW = wmma_bf16(a_cur, bw_cur, accW);
      accK = wmma_bf16(a_cur, bk_cur, accK);
      a_cur = a_nxt; bw_cur = bw_nxt; bk_cur = bk_nxt;
    }
    // x_t @ Pm^T and x_t @ P_z^T : K = 64 -> 2 k-tiles (fp32 -> bf16 on the fly)
    #pragma unroll
    for (int kt = 0; kt < D_ / 32; ++kt) {
      const int c0 = kt * 32 + abase;
      v4f f0 = *(const v4f*)&xb[arow * D_ + c0];
      v4f f1 = *(const v4f*)&xb[arow * D_ + c0 + 4];
      v4f f2 = *(const v4f*)&xb[arow * D_ + c0 + 16];
      v4f f3 = *(const v4f*)&xb[arow * D_ + c0 + 20];
      v16bf a;
      #pragma unroll
      for (int i = 0; i < 4; ++i) {
        a[i]      = (bf16)f0[i];
        a[i + 4]  = (bf16)f1[i];
        a[i + 8]  = (bf16)f2[i];
        a[i + 12] = (bf16)f3[i];
      }
      accW = wmma_bf16(a, bpm[kt], accW);
      accK = wmma_bf16(a, bpz[kt], accK);
    }

    // gated update + stream hidden to HBM (fp32, non-temporal: write-once)
    v8f vnew;
    #pragma unroll
    for (int j = 0; j < 8; ++j) {
      float z = (10.0f * DT_) / (1.0f + __expf(-(accK[j] + bzl)));
      vnew[j] = (1.0f - z) * v[j] + DT_ * (accW[j] + bvl);
      int m = j + abase;                              // C/D layout: M = j + 8*half
      __builtin_nontemporal_store(vnew[j],
          &hidden[((size_t)(b0 + m) * T_ + t) * H_ + h]);
    }
    v = vnew;

    // Publish r_{t+1} into the other buffer (no WAR: readers used rb=cur).
    #pragma unroll
    for (int j = 0; j < 8; ++j) {
      int m = j + abase;
      float r = vnew[j] > 0.0f ? vnew[j] : 0.0f;
      rbuf[nxt][m * H_ + h] = (bf16)r;
    }

#ifdef USE_ASYNC_X
    __builtin_amdgcn_s_wait_asynccnt(0);   // x_{t+1} landed in LDS
#else
    if (t + 1 < T_) xbuf[nxt][xm * D_ + xd] = xstage;
#endif
    __syncthreads();   // single barrier per step publishes rbuf[nxt] + xbuf[nxt]
    cur = nxt;
  }
}

// ---------------------------------------------------------------------------
// Head: out = hidden @ fc_w^T + fc_b.  (B*T x 512) @ (512 x 64).
// grid = 1024 blocks * 8 waves; each wave: one 16-row M tile x all 4 N tiles.
// hidden is read-once -> non-temporal loads; out is write-once -> NT stores.
// ---------------------------------------------------------------------------
__global__ __launch_bounds__(256)
void head_kernel(const float* __restrict__ hidden, const bf16* __restrict__ Fp,
                 const float* __restrict__ fcb, float* __restrict__ out)
{
  const int lane  = threadIdx.x & 31;
  const int wave  = threadIdx.x >> 5;
  const size_t row0 = ((size_t)blockIdx.x * 8 + wave) * 16;
  const int half  = lane >> 4;
  const int abase = half * 8;
  const int n     = lane & 15;
  const float* arow = hidden + (row0 + (lane & 15)) * H_;
  const v16bf* ffrag = (const v16bf*)Fp;

  v8f zero = {0.f,0.f,0.f,0.f,0.f,0.f,0.f,0.f};
  v8f acc[4] = {zero, zero, zero, zero};

  #pragma unroll 4
  for (int kt = 0; kt < H_ / 32; ++kt) {
    const int c0 = kt * 32 + abase;
    v4f a0 = __builtin_nontemporal_load((const v4f*)&arow[c0]);
    v4f a1 = __builtin_nontemporal_load((const v4f*)&arow[c0 + 4]);
    v4f a2 = __builtin_nontemporal_load((const v4f*)&arow[c0 + 16]);
    v4f a3 = __builtin_nontemporal_load((const v4f*)&arow[c0 + 20]);
    v16bf a;
    #pragma unroll
    for (int i = 0; i < 4; ++i) {
      a[i]      = (bf16)a0[i];
      a[i + 4]  = (bf16)a1[i];
      a[i + 8]  = (bf16)a2[i];
      a[i + 12] = (bf16)a3[i];
    }
    #pragma unroll
    for (int ntile = 0; ntile < 4; ++ntile)
      acc[ntile] = wmma_bf16(a, ffrag[(ntile * (H_ / 32) + kt) * 32 + lane], acc[ntile]);
  }

  #pragma unroll
  for (int ntile = 0; ntile < 4; ++ntile) {
    int c = ntile * 16 + n;
    float bias = fcb[c];
    #pragma unroll
    for (int j = 0; j < 8; ++j) {
      size_t row = row0 + (size_t)(j + abase);
      __builtin_nontemporal_store(acc[ntile][j] + bias, &out[row * C_ + c]);
    }
  }
}

// ---------------------------------------------------------------------------
// Inputs (dict order): x, W, P, b_v, b_z, K, P_z, fc_w, fc_b
// Outputs: concat(out [B,T,C], hidden [B,T,H]) fp32
// Workspace: packed bf16 weights, ~1.2 MB (L2-resident)
// ---------------------------------------------------------------------------
extern "C" void kernel_launch(void* const* d_in, const int* in_sizes, int n_in,
                              void* d_out, int out_size, void* d_ws, size_t ws_size,
                              hipStream_t stream) {
  (void)in_sizes; (void)n_in; (void)out_size; (void)ws_size;
  const float* x    = (const float*)d_in[0];
  const float* W    = (const float*)d_in[1];
  const float* P    = (const float*)d_in[2];
  const float* b_v  = (const float*)d_in[3];
  const float* b_z  = (const float*)d_in[4];
  const float* Kw   = (const float*)d_in[5];
  const float* P_z  = (const float*)d_in[6];
  const float* fc_w = (const float*)d_in[7];
  const float* fc_b = (const float*)d_in[8];

  float* out    = (float*)d_out;
  float* hidden = out + (size_t)B_ * T_ * C_;   // second tuple element, in-place

  bf16* Wp  = (bf16*)d_ws;                      // 512*512
  bf16* Kp  = Wp  + H_ * H_;                    // 512*512
  bf16* Pmp = Kp  + H_ * H_;                    // 512*64 (masked P)
  bf16* Pzp = Pmp + H_ * D_;                    // 512*64
  bf16* Fp  = Pzp + H_ * D_;                    // 64*512

  // Weight packing into WMMA B-fragment order (one-time per launch, tiny)
  pack_b_kernel<<<(H_ * H_) / 256, 256, 0, stream>>>(W,    Wp,  H_, H_, H_);
  pack_b_kernel<<<(H_ * H_) / 256, 256, 0, stream>>>(Kw,   Kp,  H_, H_, H_);
  pack_b_kernel<<<(H_ * D_) / 256, 256, 0, stream>>>(P,    Pmp, H_, D_, H_ / 2); // rows >= H/2 -> 0
  pack_b_kernel<<<(H_ * D_) / 256, 256, 0, stream>>>(P_z,  Pzp, H_, D_, H_);
  pack_b_kernel<<<(C_ * H_) / 256, 256, 0, stream>>>(fc_w, Fp,  C_, H_, C_);

  // Sequential scan: 16 WGs x 32 waves, v state in registers, r via LDS
  rnn_kernel<<<B_ / 16, 1024, 0, stream>>>(x, Wp, Kp, Pmp, Pzp, b_v, b_z, hidden);

  // Head GEMM: massively parallel WMMA
  head_kernel<<<(B_ * T_ / 16) / 8, 256, 0, stream>>>(hidden, Fp, fc_b, out);
}
